// PointNetPolylineEncoder_80118319940011
// MI455X (gfx1250) — matrix operand order repaired and checked by reading, not compiled, not run on hardware
//
#include <hip/hip_runtime.h>

#define Bb 32
#define Nn 768
#define Pp 32
#define Cc 9
#define Hh 64
#define Oo 128
#define Rr (Bb*Nn*Pp)   // 786432 rows
#define BN (Bb*Nn)      // 24576 polylines
#define EPSV 1e-5f

typedef __attribute__((ext_vector_type(16))) __bf16          v16bf;
typedef __attribute__((ext_vector_type(8)))  float           v8f;
typedef __attribute__((ext_vector_type(8)))  unsigned short  ushort8;
typedef __attribute__((ext_vector_type(4)))  float           float4v;

union FragBF { v16bf v; unsigned short u[16]; ushort8 h[2]; };
union FragF  { v8f   v; float          f[8];  };

__device__ __forceinline__ unsigned short f2bf(float f) {
  unsigned u = __builtin_bit_cast(unsigned, f);
  unsigned r = 0x7FFFu + ((u >> 16) & 1u);      // round-to-nearest-even
  return (unsigned short)((u + r) >> 16);
}
__device__ __forceinline__ float bf2f(unsigned short h) {
  unsigned u = ((unsigned)h) << 16;
  return __builtin_bit_cast(float, u);
}

// ---------------------------------------------------------------- zero stats
__global__ void k_zero(float* p, int n) {
  int i = blockIdx.x * 256 + threadIdx.x;
  if (i < n) p[i] = 0.f;
}

// ------------------------------------------- GEMM0 (K=9, VALU) + stats0 acc
__global__ void k_pre(const float* __restrict__ x, const unsigned char* __restrict__ mask,
                      const float* __restrict__ Wpre,
                      unsigned short* __restrict__ h0,
                      float* __restrict__ sum0, float* __restrict__ sq0,
                      float* __restrict__ cnt) {
  __shared__ float sW[Hh * Cc];
  __shared__ alignas(16) unsigned short tile[256 * Hh];   // 32 KB bf16 staging
  __shared__ float sM[256];
  __shared__ float part[8 * Hh];              // 4 groups x (sum,sq) x 64
  int tid = threadIdx.x;
  for (int i = tid; i < Hh * Cc; i += 256) sW[i] = Wpre[i];
  int r = blockIdx.x * 256 + tid;
  float xr[Cc];
#pragma unroll
  for (int j = 0; j < Cc; j++) xr[j] = x[(size_t)r * Cc + j];
  float m = mask[r] ? 1.f : 0.f;
  sM[tid] = m;
  __syncthreads();
#pragma unroll 4
  for (int c = 0; c < Hh; c++) {
    float a = 0.f;
#pragma unroll
    for (int j = 0; j < Cc; j++) a = __builtin_fmaf(xr[j], sW[c * Cc + j], a);
    tile[tid * Hh + c] = f2bf(a);
  }
  __syncthreads();
  // coalesced b128 writeback of the whole block tile (256*64 bf16)
  {
    ushort8* dst = (ushort8*)(h0 + (size_t)blockIdx.x * 256 * Hh);
    const ushort8* src = (const ushort8*)tile;
#pragma unroll
    for (int i = 0; i < 8; i++) dst[tid + i * 256] = src[tid + i * 256];
  }
  int ch = tid & 63, g = tid >> 6;            // 4 row-groups of 64
  float s = 0.f, q = 0.f;
  for (int rr = 0; rr < 64; rr++) {
    int row = g * 64 + rr;
    float hv = bf2f(tile[row * Hh + ch]);
    float mv = sM[row];
    s += hv * mv;
    q += hv * hv * mv;
  }
  part[(g * 2 + 0) * Hh + ch] = s;
  part[(g * 2 + 1) * Hh + ch] = q;
  __syncthreads();
  if (tid < Hh) {
    float S = 0.f, Q = 0.f;
    for (int g2 = 0; g2 < 4; g2++) { S += part[(g2 * 2) * Hh + tid]; Q += part[(g2 * 2 + 1) * Hh + tid]; }
    atomicAdd(&sum0[tid], S);
    atomicAdd(&sq0[tid], Q);
  }
  if (tid == 64) {
    float c2 = 0.f;
    for (int i = 0; i < 256; i++) c2 += sM[i];
    atomicAdd(cnt, c2);
  }
}

// ----------------------------------------------------- finalize BN stats
__global__ void k_fin(const float* sum, const float* sq, const float* cnt,
                      const float* g, const float* b,
                      float* scale, float* shift) {
  int c = threadIdx.x;
  if (c < Hh) {
    float n = cnt[0];
    float mean = sum[c] / n;
    float var = sq[c] / n - mean * mean;
    float inv = rsqrtf(var + EPSV);
    float sc = g[c] * inv;
    scale[c] = sc;
    shift[c] = b[c] - mean * sc;
  }
}

// ---- fused: BN0-apply + maxpool + concat + WMMA GEMM1 (M32 N64 K128) + stats1
__global__ void k_gemm1(const unsigned short* __restrict__ h0,
                        const unsigned char* __restrict__ mask,
                        const float* __restrict__ scale0, const float* __restrict__ shift0,
                        const float* __restrict__ W1,            // [64][128]
                        unsigned short* __restrict__ h1,
                        float* __restrict__ sum1, float* __restrict__ sq1) {
  __shared__ alignas(16) unsigned short sCat[Pp * 2 * Hh];       // 32x128 bf16 = 8 KB
  __shared__ alignas(16) unsigned short sBfrag[4 * 4 * 32 * 16]; // frag-order W1^T, 16 KB
  __shared__ alignas(16) float sM[Pp];
  __shared__ float sStat[2 * Hh];
  int tid = threadIdx.x;
  int bn = blockIdx.x;
  const unsigned short* h0p = h0 + (size_t)bn * Pp * Hh;
  __builtin_prefetch(h0p + Pp * Hh, 0, 1);       // global_prefetch_b8 next tile

  // W1[n][k] -> fragment-ordered LDS: [nt][kt][lane][e], lane's 16 elems contiguous
  for (int i = tid; i < 4 * 4 * 32 * 16; i += 256) {
    int e  = i & 15;
    int L  = (i >> 4) & 31;
    int kt = (i >> 9) & 3;
    int nt = i >> 11;
    int n  = nt * 16 + (L & 15);
    int k  = kt * 32 + ((L < 16) ? 0 : 8) + ((e < 8) ? e : e + 8);
    sBfrag[i] = f2bf(W1[n * 128 + k]);
  }
  if (tid < Pp) sM[tid] = mask[(size_t)bn * Pp + tid] ? 1.f : 0.f;
  if (tid < 2 * Hh) sStat[tid] = 0.f;
  __syncthreads();
  // BN0 + ReLU + mask -> feat cols 0..63 (one b128 load per thread)
  {
    ushort8 hv = ((const ushort8*)h0p)[tid];     // 8 channels of row p
    int p = tid >> 3, c0 = (tid & 7) * 8;
    float mv = sM[p];
#pragma unroll
    for (int j = 0; j < 8; j++) {
      int c = c0 + j;
      float v = __builtin_fmaf(bf2f(hv[j]), scale0[c], shift0[c]);
      v = fmaxf(v, 0.f) * mv;
      sCat[p * (2 * Hh) + c] = f2bf(v);
    }
  }
  __syncthreads();
  if (tid < Hh) {                                // max over P -> broadcast cols 64..127
    float mx = bf2f(sCat[tid]);
    for (int p = 1; p < Pp; p++) mx = fmaxf(mx, bf2f(sCat[p * (2 * Hh) + tid]));
    unsigned short mb = f2bf(mx);
    for (int p = 0; p < Pp; p++) sCat[p * (2 * Hh) + Hh + tid] = mb;
  }
  __syncthreads();

  int wv = tid >> 5, lane = tid & 31;
  int mtile = wv >> 2, ntile = wv & 3;           // 2x4 output tiles of 16x16
  int mrow = lane & 15;
  int abase = (lane < 16) ? 0 : 8;               // 16-bit A-layout K base
  FragF acc;
#pragma unroll
  for (int i = 0; i < 8; i++) acc.f[i] = 0.f;
#pragma unroll
  for (int kt = 0; kt < 4; kt++) {
    FragBF a, b;
    int ai = (mtile * 16 + mrow) * (2 * Hh) + kt * 32 + abase;   // 16B aligned
    a.h[0] = *(const ushort8*)&sCat[ai];
    a.h[1] = *(const ushort8*)&sCat[ai + 16];
    const ushort8* bp = (const ushort8*)&sBfrag[((ntile * 4 + kt) * 32 + lane) * 16];
    b.h[0] = bp[0];
    b.h[1] = bp[1];
    acc.v = __builtin_amdgcn_wmma_f32_16x16x32_bf16(false, a.v, false, b.v,
                                                    (short)0, acc.v, false, false);
  }
  unsigned short* h1p = h1 + (size_t)bn * Pp * Hh;
  int nc = ntile * 16 + (lane & 15);
  int mOff = mtile * 16 + ((lane < 16) ? 0 : 8);  // multiple of 8 -> 16B aligned sM reads
  float4v m0 = *(const float4v*)&sM[mOff];
  float4v m1 = *(const float4v*)&sM[mOff + 4];
  float s = 0.f, q = 0.f;                         // in-register masked partials
#pragma unroll
  for (int i = 0; i < 8; i++) {
    float v = acc.f[i];
    h1p[(mOff + i) * Hh + nc] = f2bf(v);
    float mv = (i < 4) ? m0[i & 3] : m1[i & 3];
    s = __builtin_fmaf(v, mv, s);
    q = __builtin_fmaf(v * v, mv, q);
  }
  atomicAdd(&sStat[nc], s);                       // 2 LDS atomics per lane
  atomicAdd(&sStat[Hh + nc], q);
  __syncthreads();
  if (tid < Hh)            atomicAdd(&sum1[tid], sStat[tid]);
  else if (tid < 2 * Hh)   atomicAdd(&sq1[tid - Hh], sStat[tid]);
}

// ---------- fused: BN1-apply + WMMA GEMM2 (M32 N64 K64) + stats2
__global__ void k_gemm2(const unsigned short* __restrict__ h1,
                        const unsigned char* __restrict__ mask,
                        const float* __restrict__ scale1, const float* __restrict__ shift1,
                        const float* __restrict__ W2,            // [64][64]
                        unsigned short* __restrict__ h2,
                        float* __restrict__ sum2, float* __restrict__ sq2) {
  __shared__ alignas(16) unsigned short sG[Pp * Hh];             // 4 KB
  __shared__ alignas(16) unsigned short sBfrag[4 * 2 * 32 * 16]; // frag-order W2^T, 8 KB
  __shared__ alignas(16) float sM[Pp];
  __shared__ float sStat[2 * Hh];
  int tid = threadIdx.x;
  int bn = blockIdx.x;
  const unsigned short* h1p = h1 + (size_t)bn * Pp * Hh;
  __builtin_prefetch(h1p + Pp * Hh, 0, 1);

  // W2[n][k] -> fragment-ordered LDS: [nt][kt][lane][e]
  for (int i = tid; i < 4 * 2 * 32 * 16; i += 256) {
    int e  = i & 15;
    int L  = (i >> 4) & 31;
    int kt = (i >> 9) & 1;
    int nt = i >> 10;
    int n  = nt * 16 + (L & 15);
    int k  = kt * 32 + ((L < 16) ? 0 : 8) + ((e < 8) ? e : e + 8);
    sBfrag[i] = f2bf(W2[n * 64 + k]);
  }
  if (tid < Pp) sM[tid] = mask[(size_t)bn * Pp + tid] ? 1.f : 0.f;
  if (tid < 2 * Hh) sStat[tid] = 0.f;
  __syncthreads();
  {
    ushort8 hv = ((const ushort8*)h1p)[tid];
    int p = tid >> 3, c0 = (tid & 7) * 8;
    float mv = sM[p];
    unsigned short outp[8];
#pragma unroll
    for (int j = 0; j < 8; j++) {
      int c = c0 + j;
      float v = __builtin_fmaf(bf2f(hv[j]), scale1[c], shift1[c]);
      v = fmaxf(v, 0.f) * mv;
      outp[j] = f2bf(v);
    }
    *(ushort8*)&sG[tid * 8] = *(ushort8*)outp;
  }
  __syncthreads();

  int wv = tid >> 5, lane = tid & 31;
  int mtile = wv >> 2, ntile = wv & 3;
  int mrow = lane & 15;
  int abase = (lane < 16) ? 0 : 8;
  FragF acc;
#pragma unroll
  for (int i = 0; i < 8; i++) acc.f[i] = 0.f;
#pragma unroll
  for (int kt = 0; kt < 2; kt++) {
    FragBF a, b;
    int ai = (mtile * 16 + mrow) * Hh + kt * 32 + abase;
    a.h[0] = *(const ushort8*)&sG[ai];
    a.h[1] = *(const ushort8*)&sG[ai + 16];
    const ushort8* bp = (const ushort8*)&sBfrag[((ntile * 2 + kt) * 32 + lane) * 16];
    b.h[0] = bp[0];
    b.h[1] = bp[1];
    acc.v = __builtin_amdgcn_wmma_f32_16x16x32_bf16(false, a.v, false, b.v,
                                                    (short)0, acc.v, false, false);
  }
  unsigned short* h2p = h2 + (size_t)bn * Pp * Hh;
  int nc = ntile * 16 + (lane & 15);
  int mOff = mtile * 16 + ((lane < 16) ? 0 : 8);
  float4v m0 = *(const float4v*)&sM[mOff];
  float4v m1 = *(const float4v*)&sM[mOff + 4];
  float s = 0.f, q = 0.f;
#pragma unroll
  for (int i = 0; i < 8; i++) {
    float v = acc.f[i];
    h2p[(mOff + i) * Hh + nc] = f2bf(v);
    float mv = (i < 4) ? m0[i & 3] : m1[i & 3];
    s = __builtin_fmaf(v, mv, s);
    q = __builtin_fmaf(v * v, mv, q);
  }
  atomicAdd(&sStat[nc], s);
  atomicAdd(&sStat[Hh + nc], q);
  __syncthreads();
  if (tid < Hh)            atomicAdd(&sum2[tid], sStat[tid]);
  else if (tid < 2 * Hh)   atomicAdd(&sq2[tid - Hh], sStat[tid]);
}

// ---------- BN2-apply + maxpool over P + out MLP (+valid mask)
__global__ void k_out(const unsigned short* __restrict__ h2,
                      const unsigned char* __restrict__ mask,
                      const float* __restrict__ scale2, const float* __restrict__ shift2,
                      const float* __restrict__ Wo1, const float* __restrict__ bo1,
                      const float* __restrict__ Wo2, const float* __restrict__ bo2,
                      float* __restrict__ y) {
  __shared__ float sOut[Hh];
  __shared__ float sY1[Hh];
  __shared__ float sPart[256];
  __shared__ float sM[Pp];
  int tid = threadIdx.x;
  int bn = blockIdx.x;
  const unsigned short* hp = h2 + (size_t)bn * Pp * Hh;
  if (tid < Pp) sM[tid] = mask[(size_t)bn * Pp + tid] ? 1.f : 0.f;
  __syncthreads();
  int c = tid & 63, g = tid >> 6;
  float mx = 0.f;                                // post relu*mask values are >= 0
  for (int p = g; p < Pp; p += 4) {
    float v = __builtin_fmaf(bf2f(hp[p * Hh + c]), scale2[c], shift2[c]);
    v = fmaxf(v, 0.f) * sM[p];
    mx = fmaxf(mx, v);
  }
  sPart[tid] = mx;
  __syncthreads();
  if (tid < Hh)
    sOut[tid] = fmaxf(fmaxf(sPart[tid], sPart[64 + tid]),
                      fmaxf(sPart[128 + tid], sPart[192 + tid]));
  __syncthreads();
  if (tid < Hh) {
    float a = bo1[tid];
    for (int k = 0; k < Hh; k++) a = __builtin_fmaf(sOut[k], Wo1[tid * Hh + k], a);
    sY1[tid] = fmaxf(a, 0.f);
  }
  __syncthreads();
  float vld = 0.f;
  for (int p = 0; p < Pp; p++) vld = fmaxf(vld, sM[p]);
  if (tid < Oo) {
    float a = bo2[tid];
    for (int k = 0; k < Hh; k++) a = __builtin_fmaf(sY1[k], Wo2[tid * Hh + k], a);
    y[(size_t)bn * Oo + tid] = a * vld;
  }
}

// ---------------------------------------------------------------------------
extern "C" void kernel_launch(void* const* d_in, const int* in_sizes, int n_in,
                              void* d_out, int out_size, void* d_ws, size_t ws_size,
                              hipStream_t stream) {
  const float*         x     = (const float*)d_in[0];
  const unsigned char* mask  = (const unsigned char*)d_in[1];
  const float* Wpre = (const float*)d_in[2];
  const float* gpre = (const float*)d_in[3];
  const float* bpre = (const float*)d_in[4];
  const float* W1   = (const float*)d_in[5];
  const float* g1   = (const float*)d_in[6];
  const float* b1   = (const float*)d_in[7];
  const float* W2   = (const float*)d_in[8];
  const float* g2   = (const float*)d_in[9];
  const float* b2   = (const float*)d_in[10];
  const float* Wo1  = (const float*)d_in[11];
  const float* bo1  = (const float*)d_in[12];
  const float* Wo2  = (const float*)d_in[13];
  const float* bo2  = (const float*)d_in[14];
  float* y = (float*)d_out;

  float* wsf = (float*)d_ws;
  float *sum0 = wsf + 0,   *sq0 = wsf + 64;
  float *sum1 = wsf + 128, *sq1 = wsf + 192;
  float *sum2 = wsf + 256, *sq2 = wsf + 320;
  float *cnt  = wsf + 384;
  float *scale0 = wsf + 448, *shift0 = wsf + 512;
  float *scale1 = wsf + 576, *shift1 = wsf + 640;
  float *scale2 = wsf + 704, *shift2 = wsf + 768;
  unsigned short* buf0 = (unsigned short*)((char*)d_ws + 4096);
  unsigned short* buf1 = buf0 + (size_t)Rr * Hh;

  k_zero <<<2, 256, 0, stream>>>(wsf, 448);
  k_pre  <<<Rr / 256, 256, 0, stream>>>(x, mask, Wpre, buf0, sum0, sq0, cnt);
  k_fin  <<<1, 64, 0, stream>>>(sum0, sq0, cnt, gpre, bpre, scale0, shift0);
  k_gemm1<<<BN, 256, 0, stream>>>(buf0, mask, scale0, shift0, W1, buf1, sum1, sq1);
  k_fin  <<<1, 64, 0, stream>>>(sum1, sq1, cnt, g1, b1, scale1, shift1);
  k_gemm2<<<BN, 256, 0, stream>>>(buf1, mask, scale1, shift1, W2, buf0, sum2, sq2);
  k_fin  <<<1, 64, 0, stream>>>(sum2, sq2, cnt, g2, b2, scale2, shift2);
  k_out  <<<BN, 256, 0, stream>>>(buf0, mask, scale2, shift2, Wo1, bo1, Wo2, bo2, y);
}
